// GraphJEPA_20744692040093
// MI455X (gfx1250) — compile-verified, matrix operand order checked by reading, not run on hardware
//
#include <hip/hip_runtime.h>

#define DF 128
#define BN_EPS 1e-5f

typedef __attribute__((ext_vector_type(16))) __bf16    v16bf;
typedef __attribute__((ext_vector_type(8)))  float     v8f;
typedef __attribute__((ext_vector_type(8)))  unsigned  v8u;

// round-to-nearest-even float->bf16, packed pair (lo in [15:0], hi in [31:16])
static __device__ __forceinline__ unsigned bf16pair(float lo, float hi) {
    unsigned a = __float_as_uint(lo), b = __float_as_uint(hi);
    a = (a + 0x7FFFu + ((a >> 16) & 1u)) >> 16;
    b = (b + 0x7FFFu + ((b >> 16) & 1u)) >> 16;
    return (a & 0xFFFFu) | ((b & 0xFFFFu) << 16);
}

// ---------------- degree / norm ----------------
__global__ void k_fill(float* p, float v, int n) {
    int i = blockIdx.x * blockDim.x + threadIdx.x;
    if (i < n) p[i] = v;
}

__global__ void k_deg(const long long* __restrict__ ei, int E, float* __restrict__ deg) {
    int e = blockIdx.x * blockDim.x + threadIdx.x;
    if (e < E) unsafeAtomicAdd(&deg[(int)ei[(size_t)E + e]], 1.0f);
}

__global__ void k_rsqrt(float* p, int n) {
    int i = blockIdx.x * blockDim.x + threadIdx.x;
    if (i < n) p[i] = rsqrtf(p[i]);
}

// ---------------- weight pre-pack into WMMA B-fragment layout (bf16) ----------------
// index = ((tile*4 + chunk)*32 + lane)*8 + v ; W is [128 x 128] row-major (k, col)
__global__ void k_prepack(const float* __restrict__ W1, const float* __restrict__ W2,
                          const float* __restrict__ W3, unsigned* __restrict__ wp) {
    int tid = blockIdx.x * blockDim.x + threadIdx.x;
    if (tid >= 3 * 8192) return;
    const float* W = (tid < 8192) ? W1 : ((tid < 16384) ? W2 : W3);
    int rem  = tid & 8191;
    int v    = rem & 7;
    int lane = (rem >> 3) & 31;
    int chnk = (rem >> 8) & 3;
    int tile = (rem >> 10) & 7;
    int kb   = (lane < 16) ? 0 : 8;
    int kh   = (v >= 4) ? 16 : 0;
    int k    = chnk * 32 + kb + kh + (v & 3) * 2;
    int col  = tile * 16 + (lane & 15);
    wp[tid]  = bf16pair(W[k * DF + col], W[(k + 1) * DF + col]);
}

// ---------------- WMMA GEMM: H[row0:row0+16, :] = act(X) @ W ----------------
// act = identity (bnrelu=0) or relu(x*scale+shift) (bnrelu=1), fused on LDS staging.
__global__ __launch_bounds__(256) void k_gemm(const float* __restrict__ X,
                                              const unsigned* __restrict__ Wp,
                                              const float* __restrict__ scale,
                                              const float* __restrict__ shift,
                                              int bnrelu,
                                              float* __restrict__ H, int nrows) {
    __shared__ float sx[16 * 132];   // fp32 staging: 16 rows x 128, padded stride 132
    __shared__ v8u   sa[4 * 32];     // packed A fragments: [chunk][lane], 8 dwords each

    int t = threadIdx.x;
    int lane = t & 31, wave = t >> 5;
    int row0 = blockIdx.x * 16;

    // Preload this wave's B fragments (4 chunks x 8 dwords, 32B contiguous per lane)
    // -> global_load_b128 pairs; overlaps with the LDS staging below.
    v8u bfrag[4];
    #pragma unroll
    for (int c = 0; c < 4; ++c)
        bfrag[c] = ((const v8u*)Wp)[(wave * 4 + c) * 32 + lane];

    // Stage fp32 tile (coalesced float4), fusing BN-affine + ReLU
    for (int i = t; i < 16 * 32; i += 256) {
        int r = i >> 5, c4 = i & 31;
        float4 val = make_float4(0.f, 0.f, 0.f, 0.f);
        if (row0 + r < nrows) val = ((const float4*)X)[(size_t)(row0 + r) * 32 + c4];
        if (bnrelu) {
            float4 sc = ((const float4*)scale)[c4];
            float4 sh = ((const float4*)shift)[c4];
            val.x = fmaxf(val.x * sc.x + sh.x, 0.f);
            val.y = fmaxf(val.y * sc.y + sh.y, 0.f);
            val.z = fmaxf(val.z * sc.z + sh.z, 0.f);
            val.w = fmaxf(val.w * sc.w + sh.w, 0.f);
        }
        ((float4*)sx)[r * 33 + c4] = val;
    }
    __syncthreads();

    // Pack A fragments ONCE cooperatively (1024 dwords, 4 per thread) into
    // the per-lane WMMA layout; all 8 waves then just ds_load their fragment.
    unsigned* saw = (unsigned*)sa;
    #pragma unroll
    for (int j = 0; j < 4; ++j) {
        int d  = t + j * 256;
        int c  = d >> 8, ln = (d >> 3) & 31, v = d & 7;
        int row = ln & 15;
        int k  = c * 32 + ((ln < 16) ? 0 : 8) + ((v >= 4) ? 16 : 0) + (v & 3) * 2;
        const float* rp = &sx[row * 132 + k];
        saw[d] = bf16pair(rp[0], rp[1]);
    }
    __syncthreads();

    v8f acc = {};
    #pragma unroll
    for (int c = 0; c < 4; ++c) {
        v8u au = sa[c * 32 + lane];   // two ds_load_b128
        acc = __builtin_amdgcn_wmma_f32_16x16x32_bf16(
            false, __builtin_bit_cast(v16bf, au),
            false, __builtin_bit_cast(v16bf, bfrag[c]),
            (short)0, acc, false, false);
    }

    int col   = wave * 16 + (lane & 15);
    int mbase = (lane >> 4) << 3;
    float* hp = H + (size_t)(row0 + mbase) * DF + col;
    if (row0 + 16 <= nrows) {
        // full tile (the common case): straight-line clause of stores
        #pragma unroll
        for (int v = 0; v < 8; ++v) hp[(size_t)v * DF] = acc[v];
    } else {
        #pragma unroll
        for (int v = 0; v < 8; ++v)
            if (row0 + mbase + v < nrows) hp[(size_t)v * DF] = acc[v];
    }
}

// ---------------- aggregation: AGG = bias + selfloop + scatter-add over edges ----------------
// One wave per row: row index is wave-uniform -> dinv load & norm become scalar ops.
__global__ __launch_bounds__(256) void k_agg_init(const float* __restrict__ H,
                                                  const float* __restrict__ dinv,
                                                  const float* __restrict__ bias,
                                                  float* __restrict__ AGG, int n) {
    int wave = threadIdx.x >> 5, lane = threadIdx.x & 31;
    int row = blockIdx.x * 8 + wave;
    if (row >= n) return;
    float d = dinv[row];
    float w = d * d;                         // self-loop norm
    float4 h = ((const float4*)H)[(size_t)row * 32 + lane];
    float4 b = ((const float4*)bias)[lane];
    float4 o;
    o.x = b.x + h.x * w; o.y = b.y + h.y * w;
    o.z = b.z + h.z * w; o.w = b.w + h.w * w;
    ((float4*)AGG)[(size_t)row * 32 + lane] = o;
}

// One wave per edge: endpoints + norm are wave-uniform (scalar), lanes move 16B each.
__global__ __launch_bounds__(256) void k_agg_edge(const float* __restrict__ H,
                                                  const long long* __restrict__ ei,
                                                  int E, const float* __restrict__ dinv,
                                                  float* __restrict__ AGG) {
    int wave = threadIdx.x >> 5, lane = threadIdx.x & 31;
    int e = blockIdx.x * 8 + wave;
    if (e >= E) return;
    int s = (int)ei[e];
    int d = (int)ei[(size_t)E + e];
    float nrm = dinv[s] * dinv[d];
    float4 h = ((const float4*)H)[(size_t)s * 32 + lane];
    float* p = AGG + (size_t)d * DF + lane * 4;
    unsafeAtomicAdd(p + 0, h.x * nrm);
    unsafeAtomicAdd(p + 1, h.y * nrm);
    unsafeAtomicAdd(p + 2, h.z * nrm);
    unsafeAtomicAdd(p + 3, h.w * nrm);
}

// ---------------- batchnorm ----------------
__global__ void k_bn_zero(float* sums) {
    int i = threadIdx.x;
    if (i < 256) sums[i] = 0.f;
}

__global__ void k_bn_stats(const float* __restrict__ A, int n,
                           float* __restrict__ sum, float* __restrict__ sq) {
    int f = threadIdx.x;                 // 128 threads = features
    int r0 = blockIdx.x * 128;
    int r1 = r0 + 128; if (r1 > n) r1 = n;
    float s = 0.f, s2 = 0.f;
    for (int r = r0; r < r1; ++r) {
        float v = A[(size_t)r * DF + f];
        s += v; s2 += v * v;
    }
    unsafeAtomicAdd(&sum[f], s);
    unsafeAtomicAdd(&sq[f], s2);
}

__global__ void k_bn_final(const float* __restrict__ sum, const float* __restrict__ sq,
                           const float* __restrict__ g, const float* __restrict__ be,
                           int n, float* __restrict__ scale, float* __restrict__ shift) {
    int f = threadIdx.x;
    if (f >= 128) return;
    float inv_n = 1.0f / (float)n;
    float mu  = sum[f] * inv_n;
    float var = sq[f] * inv_n - mu * mu;
    float sc  = g[f] * rsqrtf(var + BN_EPS);
    scale[f] = sc;
    shift[f] = be[f] - mu * sc;
}

// ---------------- final row L2-normalize (one wave per row) ----------------
__global__ __launch_bounds__(256) void k_rownorm(const float* __restrict__ A,
                                                 float* __restrict__ out, int n) {
    int wave = threadIdx.x >> 5, lane = threadIdx.x & 31;
    int row = blockIdx.x * 8 + wave;
    if (row >= n) return;
    float4 v = ((const float4*)A)[(size_t)row * 32 + lane];
    float s = v.x * v.x + v.y * v.y + v.z * v.z + v.w * v.w;
    #pragma unroll
    for (int m = 16; m; m >>= 1) s += __shfl_xor(s, m, 32);
    float inv = 1.0f / fmaxf(sqrtf(s), 1e-12f);
    ((float4*)out)[(size_t)row * 32 + lane] =
        make_float4(v.x * inv, v.y * inv, v.z * inv, v.w * inv);
}

// ---------------- host orchestration ----------------
extern "C" void kernel_launch(void* const* d_in, const int* in_sizes, int n_in,
                              void* d_out, int out_size, void* d_ws, size_t ws_size,
                              hipStream_t stream) {
    const float*     x   = (const float*)d_in[0];
    const long long* ei  = (const long long*)d_in[1];
    const float*     W1  = (const float*)d_in[2];
    const float*     b1  = (const float*)d_in[3];
    const float*     W2  = (const float*)d_in[4];
    const float*     b2  = (const float*)d_in[5];
    const float*     W3  = (const float*)d_in[6];
    const float*     b3  = (const float*)d_in[7];
    const float*     g1  = (const float*)d_in[8];
    const float*     be1 = (const float*)d_in[9];
    const float*     g2  = (const float*)d_in[10];
    const float*     be2 = (const float*)d_in[11];

    int N = in_sizes[0] / DF;
    int E = in_sizes[1] / 2;
    float* H = (float*)d_out;                         // reuse output buffer as H scratch

    // workspace layout (d_ws is 256B aligned; every sub-buffer stays 16B+ aligned)
    float*    AGG     = (float*)d_ws;                 // N*128
    float*    dinv    = AGG + (size_t)N * DF;         // N
    unsigned* wp      = (unsigned*)(dinv + N);        // 3*8192, 32B-aligned for v8u loads
    float*    bnsum   = (float*)(wp + 3 * 8192);      // 128
    float*    bnsq    = bnsum + 128;                  // 128  (contiguous with bnsum)
    float*    bnscale = bnsq + 128;                   // 128
    float*    bnshift = bnscale + 128;                // 128

    unsigned gN    = (unsigned)((N + 255) / 256);
    unsigned gE    = (unsigned)((E + 255) / 256);
    unsigned gN8   = (unsigned)((N + 7) / 8);         // one wave per row
    unsigned gE8   = (unsigned)((E + 7) / 8);         // one wave per edge
    unsigned gTile = (unsigned)((N + 15) / 16);
    unsigned gBN   = (unsigned)((N + 127) / 128);

    // degrees & symmetric norm
    k_fill <<<gN, 256, 0, stream>>>(dinv, 1.0f, N);          // self-loop contributes 1
    k_deg  <<<gE, 256, 0, stream>>>(ei, E, dinv);
    k_rsqrt<<<gN, 256, 0, stream>>>(dinv, N);

    // pack all 3 weight matrices to bf16 WMMA fragments
    k_prepack<<<96, 256, 0, stream>>>(W1, W2, W3, wp);

    // ---- layer 1 ----
    k_gemm    <<<gTile, 256, 0, stream>>>(x, wp, nullptr, nullptr, 0, H, N);
    k_agg_init<<<gN8, 256, 0, stream>>>(H, dinv, b1, AGG, N);
    k_agg_edge<<<gE8, 256, 0, stream>>>(H, ei, E, dinv, AGG);
    k_bn_zero <<<1, 256, 0, stream>>>(bnsum);
    k_bn_stats<<<gBN, 128, 0, stream>>>(AGG, N, bnsum, bnsq);
    k_bn_final<<<1, 128, 0, stream>>>(bnsum, bnsq, g1, be1, N, bnscale, bnshift);

    // ---- layer 2 (BN+ReLU fused into GEMM staging) ----
    k_gemm    <<<gTile, 256, 0, stream>>>(AGG, wp + 8192, bnscale, bnshift, 1, H, N);
    k_agg_init<<<gN8, 256, 0, stream>>>(H, dinv, b2, AGG, N);
    k_agg_edge<<<gE8, 256, 0, stream>>>(H, ei, E, dinv, AGG);
    k_bn_zero <<<1, 256, 0, stream>>>(bnsum);
    k_bn_stats<<<gBN, 128, 0, stream>>>(AGG, N, bnsum, bnsq);
    k_bn_final<<<1, 128, 0, stream>>>(bnsum, bnsq, g2, be2, N, bnscale, bnshift);

    // ---- layer 3 ----
    k_gemm    <<<gTile, 256, 0, stream>>>(AGG, wp + 16384, bnscale, bnshift, 1, H, N);
    k_agg_init<<<gN8, 256, 0, stream>>>(H, dinv, b3, AGG, N);
    k_agg_edge<<<gE8, 256, 0, stream>>>(H, ei, E, dinv, AGG);

    // ---- final L2 row normalize: AGG -> d_out ----
    k_rownorm <<<gN8, 256, 0, stream>>>(AGG, (float*)d_out, N);
}
